// CausalSelfAttention_60172491817880
// MI455X (gfx1250) — matrix operand order repaired and checked by reading, not compiled, not run on hardware
//
#include <hip/hip_runtime.h>
#include <hip/hip_bf16.h>

#define N_HEAD   12
#define N_EMBD   768
#define HEAD_DIM 64
#define SEQ      1024
#define BATCH    8
#define BT       (BATCH * SEQ)   // 8192 tokens
#define C3       (3 * N_EMBD)    // 2304

typedef __attribute__((ext_vector_type(16))) __bf16 v16bf;
typedef __attribute__((ext_vector_type(8)))  float  v8f;

union Frag { uint4 q[2]; v16bf v; };

__device__ __forceinline__ unsigned short f2bf(float f) {
  union { float f; unsigned u; } c; c.f = f;
  unsigned u = c.u;
  u += 0x7FFFu + ((u >> 16) & 1u);   // round-to-nearest-even
  return (unsigned short)(u >> 16);
}

__device__ __forceinline__ v8f zero8() {
  v8f z = {0.f,0.f,0.f,0.f,0.f,0.f,0.f,0.f};
  return z;
}

__device__ __forceinline__ v8f wmma_bf16(const Frag& a, const Frag& b, v8f c) {
  return __builtin_amdgcn_wmma_f32_16x16x32_bf16(false, a.v, false, b.v,
                                                 (short)0, c, false, false);
}

// ---------------------------------------------------------------------------
// Register-blocked GEMM mainloop: one wave owns an (MT*16) x (NT*16) tile.
// A: [M,K] row-major bf16.  Bt: [N,K] row-major bf16 (B transposed).
// A-frag (16x32): lane L = row L%16; K = half*8 + i (i<8), 16+half*8+(i-8).
// B-frag (32x16): lane L = col L%16; K = half*8 + i (i<8), 16+half*8+(i-8).
// K is a template constant so mi*16*K / ni*16*K fold into the b128 imm24
// offset -> 2 base pointers total, minimal VGPR/SALU addressing overhead.
// ---------------------------------------------------------------------------
template<int MT, int NT, int K>
__device__ __forceinline__ void gemm_tiles(const unsigned short* __restrict__ A,
                                           const unsigned short* __restrict__ Bt,
                                           int mBase, int nBase,
                                           v8f acc[MT][NT]) {
  const int lane = threadIdx.x & 31;
  const int lm = lane & 15, half = lane >> 4;
  const unsigned short* arow = A  + (size_t)(mBase + lm) * K + half * 8;
  const unsigned short* brow = Bt + (size_t)(nBase + lm) * K + half * 8;
#pragma unroll
  for (int mi = 0; mi < MT; ++mi)
#pragma unroll
    for (int ni = 0; ni < NT; ++ni)
      acc[mi][ni] = zero8();

  for (int k0 = 0; k0 < K; k0 += 32) {
    Frag a[MT], b[NT];
#pragma unroll
    for (int mi = 0; mi < MT; ++mi) {
      a[mi].q[0] = *(const uint4*)(arow + k0 + mi * 16 * K);
      a[mi].q[1] = *(const uint4*)(arow + k0 + mi * 16 * K + 16);
    }
#pragma unroll
    for (int ni = 0; ni < NT; ++ni) {
      b[ni].q[0] = *(const uint4*)(brow + k0 + ni * 16 * K);
      b[ni].q[1] = *(const uint4*)(brow + k0 + ni * 16 * K + 16);
    }
#pragma unroll
    for (int mi = 0; mi < MT; ++mi)
#pragma unroll
      for (int ni = 0; ni < NT; ++ni)
        acc[mi][ni] = wmma_bf16(a[mi], b[ni], acc[mi][ni]);
  }
}

// ---------------------------------------------------------------------------
__global__ void cvt_bf16(const float* __restrict__ src,
                         unsigned short* __restrict__ dst, int n) {
  int i = blockIdx.x * blockDim.x + threadIdx.x;
  if (i < n) dst[i] = f2bf(src[i]);
}

// W [K,N] row-major f32  ->  Wt [N,K] row-major bf16
__global__ void transpose_w(const float* __restrict__ W,
                            unsigned short* __restrict__ Wt, int K, int N) {
  int idx = blockIdx.x * blockDim.x + threadIdx.x;
  if (idx >= N * K) return;
  int n = idx / K, k = idx % K;
  Wt[idx] = f2bf(W[(size_t)k * N + n]);
}

// ---------------------------------------------------------------------------
// qkv = x @ W_attn + b_attn -> q [B,H,T,D], k [B,H,T,D], v [B,H,D,T] (bf16)
// One wave = 64x64 output tile.
// ---------------------------------------------------------------------------
__global__ void __launch_bounds__(32, 1)
qkv_gemm(const unsigned short* __restrict__ xb,
         const unsigned short* __restrict__ WaT,
         const float* __restrict__ bias,
         unsigned short* __restrict__ qB,
         unsigned short* __restrict__ kB,
         unsigned short* __restrict__ vT) {
  const int lane = threadIdx.x & 31;
  const int lm = lane & 15, half = lane >> 4;
  const int n0 = blockIdx.x * 64;
  const int m0 = blockIdx.y * 64;
  v8f acc[4][4];
  gemm_tiles<4, 4, N_EMBD>(xb, WaT, m0, n0, acc);

  const int sec = n0 / N_EMBD;          // uniform: 64-wide tile within section
#pragma unroll
  for (int ni = 0; ni < 4; ++ni) {
    const int n  = n0 + ni * 16 + lm;
    const float bs = bias[n];
    const int cc = n % N_EMBD;
    const int h  = cc >> 6, d = cc & 63;
#pragma unroll
    for (int mi = 0; mi < 4; ++mi) {
#pragma unroll
      for (int r = 0; r < 8; ++r) {
        const int m = m0 + mi * 16 + r + half * 8;
        const int b = m >> 10, t = m & 1023;
        const unsigned short val = f2bf(acc[mi][ni][r] + bs);
        const size_t headBase = (size_t)(b * N_HEAD + h) * (SEQ * HEAD_DIM);
        if (sec == 0)      qB[headBase + (size_t)t * HEAD_DIM + d] = val;
        else if (sec == 1) kB[headBase + (size_t)t * HEAD_DIM + d] = val;
        else               vT[headBase + (size_t)d * SEQ + t]      = val;
      }
    }
  }
}

// ---------------------------------------------------------------------------
// Fused causal flash attention, one wave = 16 query rows of one (b,h).
// K/V tiles are staged into LDS with async-to-LDS DMA (ASYNCcnt), then WMMA
// fragments are fed from LDS (ds_load_b128).
// ---------------------------------------------------------------------------
__global__ void __launch_bounds__(32, 1)
attn_kernel(const unsigned short* __restrict__ qB,
            const unsigned short* __restrict__ kB,
            const unsigned short* __restrict__ vT,
            unsigned short* __restrict__ yB) {
  __shared__ __align__(16) unsigned short kbuf[32 * 64];  // [key][d]   4KB
  __shared__ __align__(16) unsigned short vbuf[64 * 32];  // [d][key]   4KB
  __shared__ __align__(16) unsigned short pbuf[16 * 32];  // P stage    1KB
  const int lane = threadIdx.x & 31;
  const int lm = lane & 15, half = lane >> 4;
  const int bh = blockIdx.y;
  const int q0 = blockIdx.x * 16;
  const size_t headOff = (size_t)bh * (SEQ * HEAD_DIM);
  const unsigned short* Q  = qB + headOff;
  const unsigned short* Kc = kB + headOff;
  const unsigned short* V  = vT + headOff;
  const unsigned kbuf_base = (unsigned)(uintptr_t)kbuf;   // LDS byte offset
  const unsigned vbuf_base = (unsigned)(uintptr_t)vbuf;

  // Q fragments for d 0..31 and 32..63 (loaded once)
  Frag qa0, qa1;
  const unsigned short* qrow = Q + (size_t)(q0 + lm) * HEAD_DIM + half * 8;
  qa0.q[0] = *(const uint4*)(qrow);
  qa0.q[1] = *(const uint4*)(qrow + 16);
  qa1.q[0] = *(const uint4*)(qrow + 32);
  qa1.q[1] = *(const uint4*)(qrow + 48);

  v8f o[4] = {zero8(), zero8(), zero8(), zero8()};
  float m_r[8], l_r[8];
#pragma unroll
  for (int r = 0; r < 8; ++r) { m_r[r] = -1e30f; l_r[r] = 0.f; }

  const int kend = q0 + 16;                    // causal: keys <= q0+15
  for (int kb = 0; kb < kend; kb += 32) {
    // ---- async stage K tile: rows kb..kb+32 of [T,64] -> contiguous 4KB ----
    const char* kt = (const char*)(Kc + (size_t)kb * HEAD_DIM);
#pragma unroll
    for (int i = 0; i < 8; ++i) {
      unsigned ldsa = kbuf_base + i * 512 + lane * 16;
      unsigned long long ga = (unsigned long long)(uintptr_t)(kt + i * 512 + lane * 16);
      asm volatile("global_load_async_to_lds_b128 %0, %1, off"
                   :: "v"(ldsa), "v"(ga) : "memory");
    }
    // ---- async stage V tile: 64 rows of [D,T], 64B per row ----
#pragma unroll
    for (int i = 0; i < 8; ++i) {
      const int d  = i * 8 + (lane >> 2);
      const int ch = lane & 3;
      unsigned ldsa = vbuf_base + d * 64 + ch * 16;
      unsigned long long ga =
          (unsigned long long)(uintptr_t)((const char*)(V + (size_t)d * SEQ + kb) + ch * 16);
      asm volatile("global_load_async_to_lds_b128 %0, %1, off"
                   :: "v"(ldsa), "v"(ga) : "memory");
    }
    asm volatile("s_wait_asynccnt 0x0" ::: "memory");   // LDS tiles ready

    // ---- K fragments from LDS: two 16-key tiles, d-span 0..31 / 32..63 ----
    const unsigned short* krow0 = kbuf + lm * 64 + half * 8;
    const unsigned short* krow1 = kbuf + (16 + lm) * 64 + half * 8;
    Frag k0a, k0b, k1a, k1b;
    k0a.q[0] = *(const uint4*)(krow0);      k0a.q[1] = *(const uint4*)(krow0 + 16);
    k0b.q[0] = *(const uint4*)(krow0 + 32); k0b.q[1] = *(const uint4*)(krow0 + 48);
    k1a.q[0] = *(const uint4*)(krow1);      k1a.q[1] = *(const uint4*)(krow1 + 16);
    k1b.q[0] = *(const uint4*)(krow1 + 32); k1b.q[1] = *(const uint4*)(krow1 + 48);

    v8f s0 = zero8(), s1 = zero8();
    s0 = wmma_bf16(qa0, k0a, s0); s0 = wmma_bf16(qa1, k0b, s0);
    s1 = wmma_bf16(qa0, k1a, s1); s1 = wmma_bf16(qa1, k1b, s1);

    // ---- online softmax over 32-key block ----
#pragma unroll
    for (int r = 0; r < 8; ++r) {
      const int qi = q0 + r + half * 8;
      float a0 = (kb + lm      <= qi) ? s0[r] * 0.125f : -1e30f;
      float a1 = (kb + 16 + lm <= qi) ? s1[r] * 0.125f : -1e30f;
      float mx = fmaxf(a0, a1);
      mx = fmaxf(mx, __shfl_xor(mx, 1, 32));
      mx = fmaxf(mx, __shfl_xor(mx, 2, 32));
      mx = fmaxf(mx, __shfl_xor(mx, 4, 32));
      mx = fmaxf(mx, __shfl_xor(mx, 8, 32));
      const float mnew  = fmaxf(m_r[r], mx);
      const float alpha = __expf(m_r[r] - mnew);
      const float p0 = __expf(a0 - mnew);
      const float p1 = __expf(a1 - mnew);
      float sum = p0 + p1;
      sum += __shfl_xor(sum, 1, 32);
      sum += __shfl_xor(sum, 2, 32);
      sum += __shfl_xor(sum, 4, 32);
      sum += __shfl_xor(sum, 8, 32);
      l_r[r] = l_r[r] * alpha + sum;
      m_r[r] = mnew;
      o[0][r] *= alpha; o[1][r] *= alpha; o[2][r] *= alpha; o[3][r] *= alpha;
      pbuf[(r + half * 8) * 32 + lm]      = f2bf(p0);   // C-layout -> LDS tile
      pbuf[(r + half * 8) * 32 + 16 + lm] = f2bf(p1);
    }
    // same-wave LDS RAW is in-order: reload P in A-fragment layout
    Frag pf;
    const unsigned short* prow = &pbuf[lm * 32 + half * 8];
    pf.q[0] = *(const uint4*)(prow);
    pf.q[1] = *(const uint4*)(prow + 16);

    // ---- o += P @ V from LDS-staged V tile ----
#pragma unroll
    for (int nt = 0; nt < 4; ++nt) {
      Frag vf;
      const unsigned short* vrow = vbuf + (nt * 16 + lm) * 32 + half * 8;
      vf.q[0] = *(const uint4*)(vrow);
      vf.q[1] = *(const uint4*)(vrow + 16);
      o[nt] = wmma_bf16(pf, vf, o[nt]);
    }
  }

  // normalize and write y in [B,T,C] bf16
  const int b = bh / N_HEAD, h = bh % N_HEAD;
#pragma unroll
  for (int r = 0; r < 8; ++r) {
    const float inv = 1.0f / l_r[r];
    const size_t base =
        ((size_t)b * SEQ + q0 + r + half * 8) * N_EMBD + h * HEAD_DIM + lm;
    yB[base]      = f2bf(o[0][r] * inv);
    yB[base + 16] = f2bf(o[1][r] * inv);
    yB[base + 32] = f2bf(o[2][r] * inv);
    yB[base + 48] = f2bf(o[3][r] * inv);
  }
}

// ---------------------------------------------------------------------------
// out = y @ W_proj + b_proj (f32). One wave = 64x32 output tile.
// ---------------------------------------------------------------------------
__global__ void __launch_bounds__(32, 1)
proj_gemm(const unsigned short* __restrict__ yB,
          const unsigned short* __restrict__ WpT,
          const float* __restrict__ bias,
          float* __restrict__ out) {
  const int lane = threadIdx.x & 31;
  const int lm = lane & 15, half = lane >> 4;
  const int n0 = blockIdx.x * 32;
  const int m0 = blockIdx.y * 64;
  v8f acc[4][2];
  gemm_tiles<4, 2, N_EMBD>(yB, WpT, m0, n0, acc);
#pragma unroll
  for (int ni = 0; ni < 2; ++ni) {
    const int n = n0 + ni * 16 + lm;
    const float bs = bias[n];
#pragma unroll
    for (int mi = 0; mi < 4; ++mi) {
#pragma unroll
      for (int r = 0; r < 8; ++r) {
        const int m = m0 + mi * 16 + r + half * 8;
        out[(size_t)m * N_EMBD + n] = acc[mi][ni][r] + bs;
      }
    }
  }
}

// ---------------------------------------------------------------------------
extern "C" void kernel_launch(void* const* d_in, const int* in_sizes, int n_in,
                              void* d_out, int out_size, void* d_ws, size_t ws_size,
                              hipStream_t stream) {
  const float* x      = (const float*)d_in[0];
  const float* W_attn = (const float*)d_in[1];
  const float* b_attn = (const float*)d_in[2];
  const float* W_proj = (const float*)d_in[3];
  const float* b_proj = (const float*)d_in[4];
  float* out = (float*)d_out;

  // Workspace layout (bf16 halves). Total ~64.5 MB.
  unsigned short* xb  = (unsigned short*)d_ws;
  unsigned short* WaT = xb  + (size_t)BT * N_EMBD;       // [2304,768]
  unsigned short* WpT = WaT + (size_t)C3 * N_EMBD;       // [768,768]
  unsigned short* qB  = WpT + (size_t)N_EMBD * N_EMBD;   // [B,H,T,D]
  unsigned short* kB  = qB  + (size_t)BT * N_EMBD;       // [B,H,T,D]
  unsigned short* vT  = kB  + (size_t)BT * N_EMBD;       // [B,H,D,T]
  unsigned short* yB  = vT  + (size_t)BT * N_EMBD;       // [B,T,C]

  const int nx = BT * N_EMBD;
  cvt_bf16<<<(nx + 255) / 256, 256, 0, stream>>>(x, xb, nx);
  transpose_w<<<(C3 * N_EMBD + 255) / 256, 256, 0, stream>>>(W_attn, WaT, N_EMBD, C3);
  transpose_w<<<(N_EMBD * N_EMBD + 255) / 256, 256, 0, stream>>>(W_proj, WpT, N_EMBD, N_EMBD);

  qkv_gemm<<<dim3(C3 / 64, BT / 64), 32, 0, stream>>>(xb, WaT, b_attn, qB, kB, vT);
  attn_kernel<<<dim3(SEQ / 16, BATCH * N_HEAD), 32, 0, stream>>>(qB, kB, vT, yB);
  proj_gemm<<<dim3(N_EMBD / 32, BT / 64), 32, 0, stream>>>(yB, WpT, b_proj, out);
}
